// MultilayerGRU_67688684585290
// MI455X (gfx1250) — matrix operand order repaired and checked by reading, not compiled, not run on hardware
//
#include <hip/hip_runtime.h>
#include <hip/hip_bf16.h>
#include <math.h>

typedef _Float16 v16h __attribute__((ext_vector_type(16)));
typedef float    v8f  __attribute__((ext_vector_type(8)));
typedef float    v4f  __attribute__((ext_vector_type(4)));

#define B_DIM   32
#define IN_DIM  512
#define H_DIM   1024
#define OUT_DIM 512
#define SEQ     512

static __device__ __forceinline__ v8f wmma16(v16h a, v16h b, v8f c) {
  // D = A(16x32) x B(32x16) + C, f16 in / f32 accumulate
  return __builtin_amdgcn_wmma_f32_16x16x32_f16(false, a, false, b, (short)0, c, false, false);
}

// Build wave32 16-bit A fragment (16x32) from fp32 row-major source.
// p points at A[row] + kc + sel, where sel = (lane&16)?16:0.
// v16h elements 0..15 = K (kc+sel)+0..15.
static __device__ __forceinline__ v16h a_frag_f32(const float* __restrict__ p) {
  v4f f0 = *(const v4f*)(p);
  v4f f1 = *(const v4f*)(p + 4);
  v4f f2 = *(const v4f*)(p + 8);
  v4f f3 = *(const v4f*)(p + 12);
  v16h a;
#pragma unroll
  for (int i = 0; i < 4; ++i) {
    a[i]      = (_Float16)f0[i];
    a[i + 4]  = (_Float16)f1[i];
    a[i + 8]  = (_Float16)f2[i];
    a[i + 12] = (_Float16)f3[i];
  }
  return a;
}

// A fragment = (r * h) elementwise, both fp32 sources.
static __device__ __forceinline__ v16h a_frag_rh(const float* __restrict__ pr,
                                                 const float* __restrict__ ph) {
  v4f r0 = *(const v4f*)(pr),      r1 = *(const v4f*)(pr + 4);
  v4f r2 = *(const v4f*)(pr + 8), r3 = *(const v4f*)(pr + 12);
  v4f h0 = *(const v4f*)(ph),      h1 = *(const v4f*)(ph + 4);
  v4f h2 = *(const v4f*)(ph + 8), h3 = *(const v4f*)(ph + 12);
  v16h a;
#pragma unroll
  for (int i = 0; i < 4; ++i) {
    a[i]      = (_Float16)(r0[i] * h0[i]);
    a[i + 4]  = (_Float16)(r1[i] * h1[i]);
    a[i + 8]  = (_Float16)(r2[i] * h2[i]);
    a[i + 12] = (_Float16)(r3[i] * h3[i]);
  }
  return a;
}

// ---------------------------------------------------------------------------
// Phase 1: z = sigmoid(x Wxz^T + bz + h Whz^T), r likewise,
//          xg = x Wxg^T + bg   (gate = blockIdx.y in {0,1,2})
// Block: 256 threads = 8 waves = 2 m-halves x 2 n-subtiles x 2 k-slices.
// Each block covers 32 output columns; grid.x = H/32 = 32, grid.y = 3.
// KX / XSTRIDE are compile-time so the K-loops fully pipeline (const trip count).
// ---------------------------------------------------------------------------
template <int KX, long XSTRIDE>
__global__ __launch_bounds__(256)
void gru_zrx(const float* __restrict__ xbase,
             const float* __restrict__ hprev,
             const _Float16* __restrict__ Wxz, const _Float16* __restrict__ Wxr,
             const _Float16* __restrict__ Wxg,
             const _Float16* __restrict__ Whz, const _Float16* __restrict__ Whr,
             const float* __restrict__ bz, const float* __restrict__ br,
             const float* __restrict__ bg,
             float* __restrict__ zout, float* __restrict__ rout,
             float* __restrict__ gout) {
  __shared__ float red[4][32][8];
  const int tid    = threadIdx.x;
  const int w      = tid >> 5, lane = tid & 31;
  const int kslice = w & 1;
  const int nsub   = (w >> 1) & 1;
  const int mhalf  = w >> 2;
  const int gate   = blockIdx.y;
  const int n0     = blockIdx.x * 32 + nsub * 16;
  const int rowA   = mhalf * 16 + (lane & 15);
  const int selA   = (lane & 16) ? 16 : 0;
  const int kselB  = (lane & 16) ? 16 : 0;
  const int nB     = n0 + (lane & 15);
  const int kbase  = kslice * 32;

  const _Float16* Wx  = (gate == 0) ? Wxz : (gate == 1) ? Wxr : Wxg;
  const _Float16* Wh  = (gate == 0) ? Whz : Whr;
  const float* bias   = (gate == 0) ? bz : (gate == 1) ? br : bg;
  float* outp         = (gate == 0) ? zout : (gate == 1) ? rout : gout;

  v8f c = {};
  {
    const float*    xrow  = xbase + (size_t)rowA * XSTRIDE + selA + kbase;
    const _Float16* wxrow = Wx + (size_t)nB * KX + kselB + kbase;
#pragma unroll 4
    for (int i = 0; i < KX / 64; ++i) {
      v16h a = a_frag_f32(xrow + i * 64);
      v16h b = *(const v16h*)(wxrow + i * 64);
      c = wmma16(a, b, c);
    }
  }
  if (gate < 2) {
    const float*    hrow  = hprev + (size_t)rowA * H_DIM + selA + kbase;
    const _Float16* whrow = Wh + (size_t)nB * H_DIM + kselB + kbase;
#pragma unroll 4
    for (int i = 0; i < H_DIM / 64; ++i) {
      v16h a = a_frag_f32(hrow + i * 64);
      v16h b = *(const v16h*)(whrow + i * 64);
      c = wmma16(a, b, c);
    }
  }
  const int pair = w >> 1;
  if (kslice) {
#pragma unroll
    for (int i = 0; i < 8; ++i) red[pair][lane][i] = c[i];
  }
  __syncthreads();
  if (!kslice) {
    const int   n  = n0 + (lane & 15);
    const float bn = bias[n];
    const int   mb = mhalf * 16 + ((lane & 16) ? 8 : 0);
#pragma unroll
    for (int i = 0; i < 8; ++i) {
      float v = c[i] + red[pair][lane][i] + bn;
      if (gate < 2) v = 1.0f / (1.0f + __expf(-v));
      outp[(size_t)(mb + i) * H_DIM + n] = v;
    }
  }
}

// ---------------------------------------------------------------------------
// Phase 2: g = tanh(xg + (r*h) Whg^T); h_new = z*h + (1-z)*g.  grid.x = 32
// ---------------------------------------------------------------------------
__global__ __launch_bounds__(256)
void gru_g(const float* __restrict__ rbuf, const float* __restrict__ hprev,
           const _Float16* __restrict__ Whg, const float* __restrict__ xg,
           const float* __restrict__ zbuf, float* __restrict__ hnew) {
  __shared__ float red[4][32][8];
  const int tid    = threadIdx.x;
  const int w      = tid >> 5, lane = tid & 31;
  const int kslice = w & 1;
  const int nsub   = (w >> 1) & 1;
  const int mhalf  = w >> 2;
  const int n0     = blockIdx.x * 32 + nsub * 16;
  const int rowA   = mhalf * 16 + (lane & 15);
  const int selA   = (lane & 16) ? 16 : 0;
  const int kselB  = (lane & 16) ? 16 : 0;
  const int nB     = n0 + (lane & 15);
  const int kbase  = kslice * 32;

  v8f c = {};
  {
    const float*    rrow  = rbuf + (size_t)rowA * H_DIM + selA + kbase;
    const float*    hrow  = hprev + (size_t)rowA * H_DIM + selA + kbase;
    const _Float16* whrow = Whg + (size_t)nB * H_DIM + kselB + kbase;
#pragma unroll 4
    for (int i = 0; i < H_DIM / 64; ++i) {
      v16h a = a_frag_rh(rrow + i * 64, hrow + i * 64);
      v16h b = *(const v16h*)(whrow + i * 64);
      c = wmma16(a, b, c);
    }
  }
  const int pair = w >> 1;
  if (kslice) {
#pragma unroll
    for (int i = 0; i < 8; ++i) red[pair][lane][i] = c[i];
  }
  __syncthreads();
  if (!kslice) {
    const int n  = n0 + (lane & 15);
    const int mb = mhalf * 16 + ((lane & 16) ? 8 : 0);
#pragma unroll
    for (int i = 0; i < 8; ++i) {
      const size_t idx = (size_t)(mb + i) * H_DIM + n;
      float g  = tanhf(c[i] + red[pair][lane][i] + xg[idx]);
      float z  = zbuf[idx];
      float hp = hprev[idx];
      hnew[idx] = z * hp + (1.0f - z) * g;
    }
  }
}

// ---------------------------------------------------------------------------
// Output projection: y_t = h1 Why^T + bhy.  grid.x = OUT/32 = 16
// ---------------------------------------------------------------------------
__global__ __launch_bounds__(256)
void gru_y(const float* __restrict__ h1, const _Float16* __restrict__ Why,
           const float* __restrict__ bhy, float* __restrict__ outbase) {
  __shared__ float red[4][32][8];
  const int tid    = threadIdx.x;
  const int w      = tid >> 5, lane = tid & 31;
  const int kslice = w & 1;
  const int nsub   = (w >> 1) & 1;
  const int mhalf  = w >> 2;
  const int n0     = blockIdx.x * 32 + nsub * 16;
  const int rowA   = mhalf * 16 + (lane & 15);
  const int selA   = (lane & 16) ? 16 : 0;
  const int kselB  = (lane & 16) ? 16 : 0;
  const int nB     = n0 + (lane & 15);
  const int kbase  = kslice * 32;

  v8f c = {};
  {
    const float*    hrow = h1 + (size_t)rowA * H_DIM + selA + kbase;
    const _Float16* wrow = Why + (size_t)nB * H_DIM + kselB + kbase;
#pragma unroll 4
    for (int i = 0; i < H_DIM / 64; ++i) {
      v16h a = a_frag_f32(hrow + i * 64);
      v16h b = *(const v16h*)(wrow + i * 64);
      c = wmma16(a, b, c);
    }
  }
  const int pair = w >> 1;
  if (kslice) {
#pragma unroll
    for (int i = 0; i < 8; ++i) red[pair][lane][i] = c[i];
  }
  __syncthreads();
  if (!kslice) {
    const int   n  = n0 + (lane & 15);
    const float bn = bhy[n];
    const int   mb = mhalf * 16 + ((lane & 16) ? 8 : 0);
    const long  ostride = (long)SEQ * OUT_DIM;
#pragma unroll
    for (int i = 0; i < 8; ++i)
      outbase[(size_t)(mb + i) * ostride + n] = c[i] + red[pair][lane][i] + bn;
  }
}

// ---------------------------------------------------------------------------
// Utilities
// ---------------------------------------------------------------------------
__global__ void cvt_f32_f16(const float* __restrict__ src, _Float16* __restrict__ dst, int n) {
  int i = blockIdx.x * blockDim.x + threadIdx.x;
  if (i < n) dst[i] = (_Float16)src[i];
}

__global__ void zero_f32(float* p, int n) {
  int i = blockIdx.x * blockDim.x + threadIdx.x;
  if (i < n) p[i] = 0.0f;
}

__global__ void copy_hidden(const float* __restrict__ h0, const float* __restrict__ h1,
                            float* __restrict__ dst) {
  int i = blockIdx.x * blockDim.x + threadIdx.x;  // 0 .. 32767
  if (i < B_DIM * H_DIM) {
    int b = i >> 10, hh = i & (H_DIM - 1);
    dst[(size_t)(b * 2 + 0) * H_DIM + hh] = h0[i];
    dst[(size_t)(b * 2 + 1) * H_DIM + hh] = h1[i];
  }
}

// ---------------------------------------------------------------------------
extern "C" void kernel_launch(void* const* d_in, const int* in_sizes, int n_in,
                              void* d_out, int out_size, void* d_ws, size_t ws_size,
                              hipStream_t stream) {
  (void)in_sizes; (void)n_in; (void)out_size; (void)ws_size;

  // inputs in setup_inputs() order
  const float* x     = (const float*)d_in[0];
  const float* Wxz0f = (const float*)d_in[1];  const float* bxz0 = (const float*)d_in[2];
  const float* Whz0f = (const float*)d_in[3];
  const float* Wxr0f = (const float*)d_in[4];  const float* bxr0 = (const float*)d_in[5];
  const float* Whr0f = (const float*)d_in[6];
  const float* Wxg0f = (const float*)d_in[7];  const float* bxg0 = (const float*)d_in[8];
  const float* Whg0f = (const float*)d_in[9];
  const float* Wxz1f = (const float*)d_in[10]; const float* bxz1 = (const float*)d_in[11];
  const float* Whz1f = (const float*)d_in[12];
  const float* Wxr1f = (const float*)d_in[13]; const float* bxr1 = (const float*)d_in[14];
  const float* Whr1f = (const float*)d_in[15];
  const float* Wxg1f = (const float*)d_in[16]; const float* bxg1 = (const float*)d_in[17];
  const float* Whg1f = (const float*)d_in[18];
  const float* Whyf  = (const float*)d_in[19]; const float* bhy  = (const float*)d_in[20];

  // carve workspace: f16 weights then fp32 state buffers (all offsets 256B-multiple)
  char* p = (char*)d_ws;
  const size_t NXW0 = (size_t)H_DIM * IN_DIM;   // 524288
  const size_t NHW  = (size_t)H_DIM * H_DIM;    // 1048576
  const size_t NWY  = (size_t)OUT_DIM * H_DIM;  // 524288
  _Float16* wxz0 = (_Float16*)p; p += NXW0 * 2;
  _Float16* whz0 = (_Float16*)p; p += NHW  * 2;
  _Float16* wxr0 = (_Float16*)p; p += NXW0 * 2;
  _Float16* whr0 = (_Float16*)p; p += NHW  * 2;
  _Float16* wxg0 = (_Float16*)p; p += NXW0 * 2;
  _Float16* whg0 = (_Float16*)p; p += NHW  * 2;
  _Float16* wxz1 = (_Float16*)p; p += NHW  * 2;
  _Float16* whz1 = (_Float16*)p; p += NHW  * 2;
  _Float16* wxr1 = (_Float16*)p; p += NHW  * 2;
  _Float16* whr1 = (_Float16*)p; p += NHW  * 2;
  _Float16* wxg1 = (_Float16*)p; p += NHW  * 2;
  _Float16* whg1 = (_Float16*)p; p += NHW  * 2;
  _Float16* why  = (_Float16*)p; p += NWY  * 2;
  const size_t NST = (size_t)B_DIM * H_DIM;     // 32768 floats
  float* h0a = (float*)p; p += NST * 4;
  float* h0b = (float*)p; p += NST * 4;
  float* h1a = (float*)p; p += NST * 4;
  float* h1b = (float*)p; p += NST * 4;
  float* zb  = (float*)p; p += NST * 4;
  float* rb  = (float*)p; p += NST * 4;
  float* xgb = (float*)p; p += NST * 4;

  // 1) weight conversion fp32 -> f16 (once per launch; weights then live in L2)
  struct { const float* s; _Float16* d; size_t n; } cv[13] = {
    {Wxz0f, wxz0, NXW0}, {Whz0f, whz0, NHW}, {Wxr0f, wxr0, NXW0}, {Whr0f, whr0, NHW},
    {Wxg0f, wxg0, NXW0}, {Whg0f, whg0, NHW}, {Wxz1f, wxz1, NHW},  {Whz1f, whz1, NHW},
    {Wxr1f, wxr1, NHW},  {Whr1f, whr1, NHW}, {Wxg1f, wxg1, NHW},  {Whg1f, whg1, NHW},
    {Whyf,  why,  NWY}};
  for (int i = 0; i < 13; ++i)
    cvt_f32_f16<<<(int)((cv[i].n + 255) / 256), 256, 0, stream>>>(cv[i].s, cv[i].d, (int)cv[i].n);

  // 2) h0 = h1 = 0 for t=0 (ping buffer index 0)
  zero_f32<<<(int)((NST + 255) / 256), 256, 0, stream>>>(h0a, (int)NST);
  zero_f32<<<(int)((NST + 255) / 256), 256, 0, stream>>>(h1a, (int)NST);

  float* h0p[2] = {h0a, h0b};
  float* h1p[2] = {h1a, h1b};
  float* yout   = (float*)d_out;

  dim3 gridZR(H_DIM / 32, 3);
  const int gridG = H_DIM / 32;
  const int gridY = OUT_DIM / 32;

  // 3) sequential scan over timesteps (graph-captured launch chain)
  for (int t = 0; t < SEQ; ++t) {
    const int pr = t & 1, pw = pr ^ 1;
    // layer 0 (x batch-stride = S*I, Kx = IN_DIM)
    gru_zrx<IN_DIM, (long)SEQ * IN_DIM><<<gridZR, 256, 0, stream>>>(
        x + (size_t)t * IN_DIM, h0p[pr],
        wxz0, wxr0, wxg0, whz0, whr0, bxz0, bxr0, bxg0, zb, rb, xgb);
    gru_g<<<gridG, 256, 0, stream>>>(rb, h0p[pr], whg0, xgb, zb, h0p[pw]);
    // layer 1 (input = new h0, stride H, Kx = H_DIM)
    gru_zrx<H_DIM, (long)H_DIM><<<gridZR, 256, 0, stream>>>(
        h0p[pw], h1p[pr],
        wxz1, wxr1, wxg1, whz1, whr1, bxz1, bxr1, bxg1, zb, rb, xgb);
    gru_g<<<gridG, 256, 0, stream>>>(rb, h1p[pr], whg1, xgb, zb, h1p[pw]);
    // output projection
    gru_y<<<gridY, 256, 0, stream>>>(h1p[pw], why, bhy, yout + (size_t)t * OUT_DIM);
  }

  // 4) final hidden state (B, L, H); last write index after t=511 is 0
  float* hid = yout + (size_t)B_DIM * SEQ * OUT_DIM;
  copy_hidden<<<(B_DIM * H_DIM + 255) / 256, 256, 0, stream>>>(h0p[0], h1p[0], hid);
}
